// TransformerTransducerSelfAttention_14851996910068
// MI455X (gfx1250) — matrix operand order repaired and checked by reading, not compiled, not run on hardware
//
#include <hip/hip_runtime.h>
#include <hip/hip_bf16.h>

// ---------------------------------------------------------------------------
// Problem constants (match reference)
// ---------------------------------------------------------------------------
#define Bsz   2
#define Tln   2048
#define Dm    1024
#define Hh    16
#define HD    64
#define MROWS (Bsz * Tln)          // 4096 tokens total
#define SCALE 0.125f               // 64^-0.5

typedef __bf16 bf16;
typedef __attribute__((ext_vector_type(4)))  bf16  bf16x4;
typedef __attribute__((ext_vector_type(8)))  bf16  bf16x8;
typedef __attribute__((ext_vector_type(16))) bf16  bf16x16;
typedef __attribute__((ext_vector_type(8)))  float floatx8;

union ABFrag { bf16x16 v; bf16x8 h[2]; };

__device__ __forceinline__ floatx8 wmma_bf16(bf16x16 a, bf16x16 b, floatx8 c) {
    // D = A(16x32 bf16) * B(32x16 bf16) + C(16x16 f32)
    return __builtin_amdgcn_wmma_f32_16x16x32_bf16(
        /*neg_a=*/false, a, /*neg_b=*/false, b,
        /*c_mod=*/(short)0, c, /*reuse_a=*/false, /*reuse_b=*/false);
}

// A-fragment (16x32, bf16), source row-major [*, ldk]: two contiguous 8-elem runs.
// lanes 0-15: K = kk+0..15 (elems 0-15)
// lanes16-31: K = kk+16..31            (contiguous order)
__device__ __forceinline__ bf16x16 load_a(const bf16* row, int kk, int hl) {
    ABFrag f;
    f.h[0] = *(const bf16x8*)(row + kk + hl * 16);
    f.h[1] = *(const bf16x8*)(row + kk + hl * 16 + 8);
    return f.v;
}

// ---------------------------------------------------------------------------
// CDNA5 async global->LDS copy (ASYNCcnt-tracked) + wait, via inline asm
// (portable across ROCm7.2 / amdgpu-toolchain builtin-arity differences).
// Low 32 bits of a generic pointer to __shared__ are the LDS byte offset.
// ---------------------------------------------------------------------------
__device__ __forceinline__ void async_copy_b128(const bf16* gsrc, bf16* lds_dst) {
    unsigned lds_off = (unsigned)(uintptr_t)lds_dst;
    asm volatile("global_load_async_to_lds_b128 %0, %1, off"
                 :: "v"(lds_off), "v"(gsrc) : "memory");
}
__device__ __forceinline__ void wait_asynccnt_le1() {
    asm volatile("s_wait_asynccnt 0x1" ::: "memory");
}
__device__ __forceinline__ void wait_asynccnt_0() {
    asm volatile("s_wait_asynccnt 0x0" ::: "memory");
}

// ---------------------------------------------------------------------------
// 0) f32 -> bf16 conversion (X and the 4 weight matrices)
// ---------------------------------------------------------------------------
__global__ void cvt_bf16_kernel(const float* __restrict__ in, bf16* __restrict__ out, int n) {
    int i = (blockIdx.x * blockDim.x + threadIdx.x) * 4;
    if (i + 3 < n) {
        float4 v = *(const float4*)(in + i);
        bf16x4 o = { (bf16)v.x, (bf16)v.y, (bf16)v.z, (bf16)v.w };
        *(bf16x4*)(out + i) = o;
    }
}

// ---------------------------------------------------------------------------
// GEMM skeleton: block = 8 waves; block tile = 256(M) x 64(N); wave = 32x64.
// Weight tile [64n x 32k] staged in LDS via async copies, double-buffered;
// all 8 waves share it (8x global traffic reduction). A-fragments from global.
// ---------------------------------------------------------------------------
#define WT_PAD 40   // row pitch in bf16 elems (80B) -> conflict-free B-frag reads

// ---------------------------------------------------------------------------
// 1) QKV projection: y = X @ W^T + b   (z = 0:Q(scaled), 1:K, 2:V(transposed))
// ---------------------------------------------------------------------------
__global__ void __launch_bounds__(256)
qkv_proj_kernel(const bf16* __restrict__ X,
                const bf16* __restrict__ Wqb, const bf16* __restrict__ Wkb,
                const bf16* __restrict__ Wvb,
                const float* __restrict__ bq, const float* __restrict__ bk,
                const float* __restrict__ bv,
                bf16* __restrict__ Qo, bf16* __restrict__ Ko, bf16* __restrict__ VTo) {
    __shared__ __align__(16) bf16 Wt[2][64][WT_PAD];   // double-buffered weight tile

    const int tid  = threadIdx.x;
    const int lane = tid & 31;
    const int wv   = tid >> 5;
    const int hl   = lane >> 4;
    const int ln   = lane & 15;

    const int z  = blockIdx.z;
    const bf16*  W    = (z == 0) ? Wqb : (z == 1) ? Wkb : Wvb;
    const float* bias = (z == 0) ? bq  : (z == 1) ? bk  : bv;

    const int m0 = (blockIdx.y * 8 + wv) * 32;   // 32 token rows per wave
    const int n0 = blockIdx.x * 64;

    // cooperative copy mapping: 256 threads x 16B = 4KB tile (64 rows x 64B)
    const int crow = tid >> 2;             // 0..63
    const int cseg = (tid & 3) * 8;        // bf16 offset within row: 0,8,16,24
    const bf16* wsrc = W + (size_t)(n0 + crow) * Dm + cseg;

    const bf16* arow0 = X + (size_t)(m0 + ln) * Dm;
    const bf16* arow1 = X + (size_t)(m0 + 16 + ln) * Dm;

    floatx8 acc[8];
#pragma unroll
    for (int i = 0; i < 8; ++i) acc[i] = (floatx8){0.f,0.f,0.f,0.f,0.f,0.f,0.f,0.f};

    // prologue: stage K-step 0
    async_copy_b128(wsrc, &Wt[0][crow][cseg]);

    const int NIT = Dm / 32;
    for (int it = 0; it < NIT; ++it) {
        const int kk  = it * 32;
        const int cur = it & 1;
        if (it + 1 < NIT) {
            async_copy_b128(wsrc + kk + 32, &Wt[cur ^ 1][crow][cseg]);
            wait_asynccnt_le1();           // current buffer's copy complete
        } else {
            wait_asynccnt_0();
        }
        __syncthreads();                   // all waves' copies visible

        bf16x16 a0 = load_a(arow0, kk, hl);
        bf16x16 a1 = load_a(arow1, kk, hl);
#pragma unroll
        for (int sub = 0; sub < 4; ++sub) {
            ABFrag bfr;                    // B-frag: N=ln, K=hl*16+e (from LDS)
            bfr.h[0] = *(const bf16x8*)(&Wt[cur][sub * 16 + ln][hl * 16]);
            bfr.h[1] = *(const bf16x8*)(&Wt[cur][sub * 16 + ln][hl * 16 + 8]);
            acc[sub]     = wmma_bf16(a0, bfr.v, acc[sub]);
            acc[4 + sub] = wmma_bf16(a1, bfr.v, acc[4 + sub]);
        }
        __syncthreads();                   // done reading before next overwrite
    }

#pragma unroll
    for (int sub = 0; sub < 4; ++sub) {
        const int n  = n0 + sub * 16 + ln;
        const int hh = n >> 6, d = n & 63;
        const float bval = bias[n];
#pragma unroll
        for (int half = 0; half < 2; ++half) {
#pragma unroll
            for (int r = 0; r < 8; ++r) {
                const int m = m0 + half * 16 + r + 8 * hl;   // C layout M
                const int b = m >> 11, t = m & (Tln - 1);
                float val = acc[half * 4 + sub][r] + bval;
                if (z == 0) {
                    Qo[(((size_t)b * Hh + hh) * Tln + t) * HD + d] = (bf16)(val * SCALE);
                } else if (z == 1) {
                    Ko[(((size_t)b * Hh + hh) * Tln + t) * HD + d] = (bf16)val;
                } else {
                    VTo[(((size_t)b * Hh + hh) * HD + d) * Tln + t] = (bf16)val;
                }
            }
        }
    }
}

// ---------------------------------------------------------------------------
// 2) Flash attention: per wave = one (b,h) x 16-query tile; stream keys by 32.
//    scores: 4 WMMA; online softmax (reduce over 16-lane groups); P -> LDS
//    (C-layout -> A-layout transpose); PV: 4 WMMA into 16x64 f32 accumulator.
//    Q/K/VT working set (24MB bf16) is L2-resident on MI455X (192MB L2).
// ---------------------------------------------------------------------------
__global__ void __launch_bounds__(256)
flash_attn_kernel(const bf16* __restrict__ Q, const bf16* __restrict__ K,
                  const bf16* __restrict__ VT, const float* __restrict__ mask,
                  bf16* __restrict__ ctx) {
    __shared__ __align__(16) bf16 Plds[8][16][32];   // per-wave 16x32 prob tile

    const int lane = threadIdx.x & 31;
    const int wv   = threadIdx.x >> 5;
    const int hl   = lane >> 4;
    const int ln   = lane & 15;

    const int bh = blockIdx.x;              // 0..31
    const int b  = bh / Hh, h = bh % Hh;
    const int q0 = (blockIdx.y * 8 + wv) * 16;

    const bf16*  qbase = Q  + (((size_t)bh * Tln) + q0 + ln) * HD;
    const bf16*  kbase = K  + ((size_t)bh * Tln) * HD;
    const bf16*  vbase = VT + ((size_t)bh * HD) * Tln;
    const float* mbase = mask + (size_t)b * Tln * Tln;

    const bf16x16 qa0 = load_a(qbase, 0, hl);
    const bf16x16 qa1 = load_a(qbase, 32, hl);

    floatx8 acc[4];
#pragma unroll
    for (int i = 0; i < 4; ++i) acc[i] = (floatx8){0.f,0.f,0.f,0.f,0.f,0.f,0.f,0.f};
    float mrow[8], lrow[8];
#pragma unroll
    for (int r = 0; r < 8; ++r) { mrow[r] = -1.0e30f; lrow[r] = 0.f; }

    for (int kk = 0; kk < Tln; kk += 32) {
        floatx8 s0 = {0.f,0.f,0.f,0.f,0.f,0.f,0.f,0.f};
        floatx8 s1 = {0.f,0.f,0.f,0.f,0.f,0.f,0.f,0.f};
        const bf16* kr0 = kbase + (size_t)(kk + ln) * HD;
        const bf16* kr1 = kbase + (size_t)(kk + 16 + ln) * HD;
        bf16x16 b00 = *(const bf16x16*)(kr0 + hl * 16);
        bf16x16 b01 = *(const bf16x16*)(kr0 + 32 + hl * 16);
        bf16x16 b10 = *(const bf16x16*)(kr1 + hl * 16);
        bf16x16 b11 = *(const bf16x16*)(kr1 + 32 + hl * 16);
        s0 = wmma_bf16(qa0, b00, s0);
        s0 = wmma_bf16(qa1, b01, s0);
        s1 = wmma_bf16(qa0, b10, s1);
        s1 = wmma_bf16(qa1, b11, s1);

        float fsc[8];
#pragma unroll
        for (int r = 0; r < 8; ++r) {
            const int qrow = q0 + r + 8 * hl;
            float v0 = s0[r] + mbase[(size_t)qrow * Tln + kk + ln];
            float v1 = s1[r] + mbase[(size_t)qrow * Tln + kk + 16 + ln];
            float mx = fmaxf(v0, v1);
            mx = fmaxf(mx, __shfl_xor(mx, 8, 32));
            mx = fmaxf(mx, __shfl_xor(mx, 4, 32));
            mx = fmaxf(mx, __shfl_xor(mx, 2, 32));
            mx = fmaxf(mx, __shfl_xor(mx, 1, 32));
            const float nm = fmaxf(mrow[r], mx);
            const float sc = __expf(mrow[r] - nm);
            const float p0 = __expf(v0 - nm);
            const float p1 = __expf(v1 - nm);
            float rs = p0 + p1;
            rs += __shfl_xor(rs, 8, 32);
            rs += __shfl_xor(rs, 4, 32);
            rs += __shfl_xor(rs, 2, 32);
            rs += __shfl_xor(rs, 1, 32);
            lrow[r] = lrow[r] * sc + rs;
            mrow[r] = nm;
            fsc[r]  = sc;
            Plds[wv][r + 8 * hl][ln]      = (bf16)p0;
            Plds[wv][r + 8 * hl][16 + ln] = (bf16)p1;
        }
#pragma unroll
        for (int j = 0; j < 4; ++j)
#pragma unroll
            for (int r = 0; r < 8; ++r) acc[j][r] *= fsc[r];

        // P: C-layout -> A-layout via LDS (same-wave DS ops are in-order)
        ABFrag pf;
        pf.h[0] = *(const bf16x8*)(&Plds[wv][ln][hl * 16]);
        pf.h[1] = *(const bf16x8*)(&Plds[wv][ln][hl * 16 + 8]);

#pragma unroll
        for (int j = 0; j < 4; ++j) {
            bf16x16 vb = *(const bf16x16*)(vbase + (size_t)(j * 16 + ln) * Tln + kk + hl * 16);
            acc[j] = wmma_bf16(pf.v, vb, acc[j]);
        }
    }

#pragma unroll
    for (int j = 0; j < 4; ++j) {
        const int d = h * HD + j * 16 + ln;
#pragma unroll
        for (int r = 0; r < 8; ++r) {
            const int t = q0 + r + 8 * hl;
            const float val = acc[j][r] / lrow[r];
            ctx[((size_t)b * Tln + t) * Dm + d] = (bf16)val;
        }
    }
}

// ---------------------------------------------------------------------------
// 3) Output projection: out = ctx @ Wo^T + bo (f32 out), same LDS-staged GEMM
// ---------------------------------------------------------------------------
__global__ void __launch_bounds__(256)
out_proj_kernel(const bf16* __restrict__ Ctx, const bf16* __restrict__ Wob,
                const float* __restrict__ bo, float* __restrict__ out) {
    __shared__ __align__(16) bf16 Wt[2][64][WT_PAD];

    const int tid  = threadIdx.x;
    const int lane = tid & 31;
    const int wv   = tid >> 5;
    const int hl   = lane >> 4;
    const int ln   = lane & 15;

    const int m0 = (blockIdx.y * 8 + wv) * 32;
    const int n0 = blockIdx.x * 64;

    const int crow = tid >> 2;
    const int cseg = (tid & 3) * 8;
    const bf16* wsrc = Wob + (size_t)(n0 + crow) * Dm + cseg;

    const bf16* arow0 = Ctx + (size_t)(m0 + ln) * Dm;
    const bf16* arow1 = Ctx + (size_t)(m0 + 16 + ln) * Dm;

    floatx8 acc[8];
#pragma unroll
    for (int i = 0; i < 8; ++i) acc[i] = (floatx8){0.f,0.f,0.f,0.f,0.f,0.f,0.f,0.f};

    async_copy_b128(wsrc, &Wt[0][crow][cseg]);

    const int NIT = Dm / 32;
    for (int it = 0; it < NIT; ++it) {
        const int kk  = it * 32;
        const int cur = it & 1;
        if (it + 1 < NIT) {
            async_copy_b128(wsrc + kk + 32, &Wt[cur ^ 1][crow][cseg]);
            wait_asynccnt_le1();
        } else {
            wait_asynccnt_0();
        }
        __syncthreads();

        bf16x16 a0 = load_a(arow0, kk, hl);
        bf16x16 a1 = load_a(arow1, kk, hl);
#pragma unroll
        for (int sub = 0; sub < 4; ++sub) {
            ABFrag bfr;
            bfr.h[0] = *(const bf16x8*)(&Wt[cur][sub * 16 + ln][hl * 16]);
            bfr.h[1] = *(const bf16x8*)(&Wt[cur][sub * 16 + ln][hl * 16 + 8]);
            acc[sub]     = wmma_bf16(a0, bfr.v, acc[sub]);
            acc[4 + sub] = wmma_bf16(a1, bfr.v, acc[4 + sub]);
        }
        __syncthreads();
    }

#pragma unroll
    for (int sub = 0; sub < 4; ++sub) {
        const int n = n0 + sub * 16 + ln;
        const float bval = bo[n];
#pragma unroll
        for (int half = 0; half < 2; ++half) {
#pragma unroll
            for (int r = 0; r < 8; ++r) {
                const int m = m0 + half * 16 + r + 8 * hl;
                out[(size_t)m * Dm + n] = acc[half * 4 + sub][r] + bval;
            }
        }
    }
}

// ---------------------------------------------------------------------------
// Launcher
// ---------------------------------------------------------------------------
extern "C" void kernel_launch(void* const* d_in, const int* in_sizes, int n_in,
                              void* d_out, int out_size, void* d_ws, size_t ws_size,
                              hipStream_t stream) {
    const float* hs   = (const float*)d_in[0];   // [B,T,D]
    const float* mask = (const float*)d_in[1];   // [B,1,T,T]
    const float* Wq   = (const float*)d_in[2];
    const float* bq   = (const float*)d_in[3];
    const float* Wk   = (const float*)d_in[4];
    const float* bk   = (const float*)d_in[5];
    const float* Wv   = (const float*)d_in[6];
    const float* bv   = (const float*)d_in[7];
    const float* Wo   = (const float*)d_in[8];
    const float* bo   = (const float*)d_in[9];
    float* out = (float*)d_out;

    // Workspace layout (bf16 elements): 24M elems = 48 MB total
    const size_t XN = (size_t)MROWS * Dm;        // 4M
    const size_t WN = (size_t)Dm * Dm;           // 1M
    bf16* Xb   = (bf16*)d_ws;
    bf16* Wqb  = Xb + XN;
    bf16* Wkb  = Wqb + WN;
    bf16* Wvb  = Wkb + WN;
    bf16* Wob  = Wvb + WN;
    bf16* Qb   = Wob + WN;                       // [B,H,T,HD]
    bf16* Kb   = Qb + XN;                        // [B,H,T,HD]
    bf16* VTb  = Kb + XN;                        // [B,H,HD,T]
    bf16* Ctxb = VTb + XN;                       // [B,T,D]

    cvt_bf16_kernel<<<dim3((unsigned)(XN / 1024)), 256, 0, stream>>>(hs, Xb, (int)XN);
    cvt_bf16_kernel<<<dim3((unsigned)(WN / 1024)), 256, 0, stream>>>(Wq, Wqb, (int)WN);
    cvt_bf16_kernel<<<dim3((unsigned)(WN / 1024)), 256, 0, stream>>>(Wk, Wkb, (int)WN);
    cvt_bf16_kernel<<<dim3((unsigned)(WN / 1024)), 256, 0, stream>>>(Wv, Wvb, (int)WN);
    cvt_bf16_kernel<<<dim3((unsigned)(WN / 1024)), 256, 0, stream>>>(Wo, Wob, (int)WN);

    // 1) QKV projections: block tile 256(M) x 64(N), z = Q / K / V
    qkv_proj_kernel<<<dim3(Dm / 64, MROWS / 256, 3), 256, 0, stream>>>(
        Xb, Wqb, Wkb, Wvb, bq, bk, bv, Qb, Kb, VTb);

    // 2) Flash attention: grid = (B*H, T / (16 q-rows * 8 waves))
    flash_attn_kernel<<<dim3(Bsz * Hh, Tln / 128), 256, 0, stream>>>(
        Qb, Kb, VTb, mask, Ctxb);

    // 3) Output projection
    out_proj_kernel<<<dim3(Dm / 64, MROWS / 256), 256, 0, stream>>>(
        Ctxb, Wob, bo, out);

    (void)in_sizes; (void)n_in; (void)out_size; (void)ws_size;
}
